// MyConditionalGlobalAttention_21449066676449
// MI455X (gfx1250) — matrix operand order repaired and checked by reading, not compiled, not run on hardware
//
#include <hip/hip_runtime.h>
#include <hip/hip_bf16.h>

// ---------------------------------------------------------------------------
// MI455X (gfx1250) conditional global attention pooling.
// bf16 WMMA (v_wmma_f32_16x16x32_bf16) for all 128x128 GEMMs, fp32 everywhere
// else. Tile = 128 rows x 128 cols per 256-thread block (8 wave32s, each wave
// owns a 16-row stripe and 8 accumulator tiles).
// ---------------------------------------------------------------------------

#define NN     500000
#define BB     8192
#define CC     128
#define PITCH  136            // LDS row pitch in bf16 elems (pad to avoid bank conflicts)
#define EPSF   1e-16f
#define NTILES ((NN + 127) / 128)

typedef __bf16 bf16;
typedef __attribute__((ext_vector_type(16))) __bf16 bf16x16;
typedef __attribute__((ext_vector_type(8)))  float  f32x8;

// ---------------- helpers ----------------

__device__ __forceinline__ void zero_acc(f32x8 acc[8]) {
#pragma unroll
  for (int i = 0; i < 8; ++i)
#pragma unroll
    for (int j = 0; j < 8; ++j) acc[i][j] = 0.f;
}

// Load a 128x128 fp32 tile (rows row0..row0+127, zero-padded past nrows) into
// LDS as bf16, row-major with pitch PITCH. 256 threads, float4-coalesced.
__device__ __forceinline__ void load_tile_bf16(const float* __restrict__ src,
                                               int row0, int nrows,
                                               bf16* __restrict__ dst) {
#pragma unroll
  for (int i = 0; i < 16; ++i) {
    int idx = i * 256 + threadIdx.x;       // 4096 float4 chunks
    int row = idx >> 5;                    // 32 float4 per row
    int c4  = (idx & 31) << 2;
    float4 v = make_float4(0.f, 0.f, 0.f, 0.f);
    int grow = row0 + row;
    if (grow < nrows) v = ((const float4*)src)[(size_t)grow * 32 + (idx & 31)];
    bf16* d = dst + row * PITCH + c4;
    d[0] = (bf16)v.x; d[1] = (bf16)v.y; d[2] = (bf16)v.z; d[3] = (bf16)v.w;
  }
}

// One 128x128x128 GEMM tile: acc += A(LDS bf16, pitch PITCH) @ W(frag layout).
// A-frag (16-bit 16x32): lane l holds row (l&15); VGPR elems 0..7 = K klo..klo+7,
// elems 8..15 = K klo+16..klo+23, klo = (l>>4)*8.
// B-frag (pre-swizzled by k_wconv): 32 contiguous bytes per lane.
__device__ __forceinline__ void gemm_tile(const bf16* __restrict__ A,
                                          const bf16* __restrict__ Wf,
                                          f32x8 acc[8]) {
  const int l   = threadIdx.x & 31;
  const int wv  = threadIdx.x >> 5;
  const int row = (wv << 4) + (l & 15);
  const int klo = (l >> 4) << 3;
#pragma unroll
  for (int kb = 0; kb < 4; ++kb) {
    bf16x16 a;
    ((uint4*)&a)[0] = *(const uint4*)(A + row * PITCH + kb * 32 + klo);
    ((uint4*)&a)[1] = *(const uint4*)(A + row * PITCH + kb * 32 + 16 + klo);
#pragma unroll
    for (int nb = 0; nb < 8; ++nb) {
      const bf16x16 b = *(const bf16x16*)(Wf + (size_t)(((kb << 3) + nb) * 32 + l) * 16);
      acc[nb] = __builtin_amdgcn_wmma_f32_16x16x32_bf16(
          false, a, false, b, (short)0, acc[nb], false, false);
    }
  }
}

// Store relu(acc + bias) to LDS as bf16 (C/D layout: VGPR j -> row j (+8 for
// upper half-wave), col = lane&15 within each 16-col block).
__device__ __forceinline__ void store_relu_bf16(const f32x8 acc[8],
                                                const float* __restrict__ bias,
                                                bf16* __restrict__ dst) {
  const int l  = threadIdx.x & 31;
  const int wv = threadIdx.x >> 5;
  const int r0 = (wv << 4) + ((l >> 4) << 3);
  const int cl = l & 15;
#pragma unroll
  for (int nb = 0; nb < 8; ++nb) {
    int col = (nb << 4) + cl;
    float bc = bias[col];
#pragma unroll
    for (int j = 0; j < 8; ++j) {
      float v = acc[nb][j] + bc;
      v = v > 0.f ? v : 0.f;
      dst[(r0 + j) * PITCH + col] = (bf16)v;
    }
  }
}

__device__ __forceinline__ unsigned enc_f32(float f) {
  unsigned u = __float_as_uint(f);
  return (u & 0x80000000u) ? ~u : (u | 0x80000000u);
}
__device__ __forceinline__ float dec_f32(unsigned u) {
  unsigned bits = (u & 0x80000000u) ? (u & 0x7fffffffu) : ~u;
  return __uint_as_float(bits);
}

// ---------------- kernels ----------------

__global__ __launch_bounds__(256) void k_init(float* out, unsigned* menc, float* s) {
  int i = blockIdx.x * 256 + threadIdx.x;
  if (i < BB * CC) out[i] = 0.f;
  if (i < BB) { menc[i] = 0u; s[i] = 0.f; }
}

// Repack 5 fp32 128x128 weight matrices into bf16 WMMA B-fragment order:
// frag (kb,nb), lane l: 16 bf16 = W[kb*32 + (l>>4)*8 + (t&7) + (t>>3)*16][nb*16 + (l&15)].
__global__ __launch_bounds__(256) void k_wconv(const float* w0, const float* w1,
                                               const float* w2, const float* w3,
                                               const float* w4, bf16* wb) {
  const float* Ws[5] = {w0, w1, w2, w3, w4};
  const float* W = Ws[blockIdx.x];
  bf16* dst = wb + (size_t)blockIdx.x * 16384;
  for (int slot = threadIdx.x; slot < 1024; slot += 256) {
    int f = slot >> 5, l = slot & 31;
    int kb = f >> 3, nb = f & 7;
    int n = (nb << 4) + (l & 15);
    int kbase = (kb << 5) + ((l >> 4) << 3);
    bf16* d = dst + (size_t)slot * 16;
#pragma unroll
    for (int t = 0; t < 16; ++t) d[t] = (bf16)W[(size_t)(kbase + (t & 7) + ((t >> 3) << 4)) * CC + n];
  }
}

// q = relu(u @ qw1 + qb1) @ qw2 + qb2     [B, C]
__global__ __launch_bounds__(256) void k_qmlp(const float* __restrict__ u,
                                              const float* __restrict__ qb1,
                                              const float* __restrict__ qb2,
                                              const bf16* __restrict__ wf1,
                                              const bf16* __restrict__ wf2,
                                              float* __restrict__ q) {
  __shared__ bf16 A0[128 * PITCH];
  __shared__ bf16 A1[128 * PITCH];
  const int row0 = blockIdx.x * 128;
  load_tile_bf16(u, row0, BB, A0);
  __syncthreads();
  f32x8 acc[8];
  zero_acc(acc);
  gemm_tile(A0, wf1, acc);
  store_relu_bf16(acc, qb1, A1);
  __syncthreads();
  zero_acc(acc);
  gemm_tile(A1, wf2, acc);
  const int l = threadIdx.x & 31, wv = threadIdx.x >> 5;
  const int r0 = (wv << 4) + ((l >> 4) << 3), cl = l & 15;
#pragma unroll
  for (int nb = 0; nb < 8; ++nb) {
    int col = (nb << 4) + cl;
    float bc = qb2[col];
#pragma unroll
    for (int j = 0; j < 8; ++j)
      q[(size_t)(row0 + r0 + j) * CC + col] = acc[nb][j] + bc;
  }
}

// h = mlp(x); g = q[batch]*h; gate = relu(g@gw1+gb1)@gw2+gb2; segment atomicMax.
__global__ __launch_bounds__(256) void k_gate(const float* __restrict__ x,
                                              const int* __restrict__ batch,
                                              const float* __restrict__ q,
                                              const float* __restrict__ nb1,
                                              const float* __restrict__ nb2,
                                              const float* __restrict__ gb1,
                                              const float* __restrict__ gw2,
                                              const float* __restrict__ gb2,
                                              const bf16* __restrict__ wfn1,
                                              const bf16* __restrict__ wfn2,
                                              const bf16* __restrict__ wfg1,
                                              float* __restrict__ gate,
                                              unsigned* __restrict__ menc) {
  __shared__ bf16 A0[128 * PITCH];
  __shared__ bf16 A1[128 * PITCH];
  __shared__ int  ib[128];
  const int row0 = blockIdx.x * 128;
  if (threadIdx.x < 128) {
    int r = row0 + threadIdx.x;
    ib[threadIdx.x] = batch[r < NN ? r : (NN - 1)];
  }
  load_tile_bf16(x, row0, NN, A0);
  __syncthreads();

  f32x8 acc[8];
  zero_acc(acc);
  gemm_tile(A0, wfn1, acc);              // stage 1: x @ nw1
  store_relu_bf16(acc, nb1, A1);
  __syncthreads();
  zero_acc(acc);
  gemm_tile(A1, wfn2, acc);              // stage 2: h = . @ nw2 + nb2

  const int l = threadIdx.x & 31, wv = threadIdx.x >> 5;
  const int r0 = (wv << 4) + ((l >> 4) << 3), cl = l & 15;
  {                                      // g = h * q[batch] -> A0 (bf16)
#pragma unroll
    for (int nb = 0; nb < 8; ++nb) {
      int col = (nb << 4) + cl;
      float bc = nb2[col];
#pragma unroll
      for (int j = 0; j < 8; ++j) {
        int lr = r0 + j;
        float h  = acc[nb][j] + bc;
        float qv = q[(size_t)ib[lr] * CC + col];
        A0[lr * PITCH + col] = (bf16)(h * qv);
      }
    }
  }
  __syncthreads();
  zero_acc(acc);
  gemm_tile(A0, wfg1, acc);              // stage 3: g @ gw1

  // in-register 128->1 matvec: relu(acc+gb1) . gw2, half-wave shfl reduction
  float w2v[8], b1v[8];
#pragma unroll
  for (int nb = 0; nb < 8; ++nb) {
    int col = (nb << 4) + cl;
    w2v[nb] = gw2[col];
    b1v[nb] = gb1[col];
  }
  float t[8];
#pragma unroll
  for (int j = 0; j < 8; ++j) {
    float sum = 0.f;
#pragma unroll
    for (int nb = 0; nb < 8; ++nb) {
      float v = acc[nb][j] + b1v[nb];
      v = v > 0.f ? v : 0.f;
      sum += v * w2v[nb];
    }
    t[j] = sum;
  }
#pragma unroll
  for (int m = 1; m < 16; m <<= 1)
#pragma unroll
    for (int j = 0; j < 8; ++j) t[j] += __shfl_xor(t[j], m, 32);
  if (cl == 0) {
    float b2 = gb2[0];
#pragma unroll
    for (int j = 0; j < 8; ++j) {
      int lr = r0 + j, r = row0 + lr;
      if (r < NN) {
        float g = t[j] + b2;
        gate[r] = g;
        atomicMax(&menc[ib[lr]], enc_f32(g));
      }
    }
  }
}

// e = exp(gate - m[batch]) (in place over gate buffer); s[b] += e
__global__ __launch_bounds__(256) void k_soft(const int* __restrict__ batch,
                                              float* __restrict__ ge,
                                              const unsigned* __restrict__ menc,
                                              float* __restrict__ s) {
  int r = blockIdx.x * 256 + threadIdx.x;
  if (r >= NN) return;
  int b = batch[r];
  float m = dec_f32(menc[b]);
  float e = expf(ge[r] - m);
  ge[r] = e;
  atomicAdd(&s[b], e);
}

// recompute h; out[b] += (e/(s[b]+eps)) * h with sorted-run accumulation
__global__ __launch_bounds__(256) void k_pool(const float* __restrict__ x,
                                              const int* __restrict__ batch,
                                              const float* __restrict__ nb1,
                                              const float* __restrict__ nb2,
                                              const bf16* __restrict__ wfn1,
                                              const bf16* __restrict__ wfn2,
                                              const float* __restrict__ e,
                                              const float* __restrict__ s,
                                              float* __restrict__ out) {
  __shared__ bf16  A0[128 * PITCH];
  __shared__ bf16  A1[128 * PITCH];
  __shared__ int   ib[128];
  __shared__ float alpha[128];
  const int row0 = blockIdx.x * 128;
  if (threadIdx.x < 128) {
    int r = row0 + threadIdx.x;
    int b = batch[r < NN ? r : (NN - 1)];
    ib[threadIdx.x] = b;
    alpha[threadIdx.x] = (r < NN) ? e[r] / (s[b] + EPSF) : 0.f;
  }
  load_tile_bf16(x, row0, NN, A0);
  __syncthreads();

  f32x8 acc[8];
  zero_acc(acc);
  gemm_tile(A0, wfn1, acc);
  store_relu_bf16(acc, nb1, A1);
  __syncthreads();
  zero_acc(acc);
  gemm_tile(A1, wfn2, acc);              // h (pre-bias) in accumulators

  const int l = threadIdx.x & 31, wv = threadIdx.x >> 5;
  const int r0 = (wv << 4) + ((l >> 4) << 3), cl = l & 15;
#pragma unroll
  for (int nb = 0; nb < 8; ++nb) {
    int col = (nb << 4) + cl;
    float bc = nb2[col];
    float runsum = 0.f;
    int curb = ib[r0];
#pragma unroll
    for (int j = 0; j < 8; ++j) {
      int lr = r0 + j;
      int bj = ib[lr];
      if (bj != curb) {
        atomicAdd(&out[(size_t)curb * CC + col], runsum);
        runsum = 0.f;
        curb = bj;
      }
      runsum += (acc[nb][j] + bc) * alpha[lr];
    }
    atomicAdd(&out[(size_t)curb * CC + col], runsum);
  }
}

// ---------------- host ----------------

extern "C" void kernel_launch(void* const* d_in, const int* in_sizes, int n_in,
                              void* d_out, int out_size, void* d_ws, size_t ws_size,
                              hipStream_t stream) {
  const float* x    = (const float*)d_in[0];
  const float* u    = (const float*)d_in[1];
  const int*   batch= (const int*)d_in[2];
  const float* gw1  = (const float*)d_in[4];
  const float* gb1  = (const float*)d_in[5];
  const float* gw2  = (const float*)d_in[6];
  const float* gb2  = (const float*)d_in[7];
  const float* nw1  = (const float*)d_in[8];
  const float* nb1  = (const float*)d_in[9];
  const float* nw2  = (const float*)d_in[10];
  const float* nb2  = (const float*)d_in[11];
  const float* qw1  = (const float*)d_in[12];
  const float* qb1  = (const float*)d_in[13];
  const float* qw2  = (const float*)d_in[14];
  const float* qb2  = (const float*)d_in[15];
  float* out = (float*)d_out;

  // workspace layout (bytes), ~6.43 MB total
  char* ws = (char*)d_ws;
  float*    q    = (float*)(ws + 0);            // B*C fp32        (4,194,304)
  float*    ge   = (float*)(ws + 4194304);      // gate then e, N  (2,000,000)
  unsigned* menc = (unsigned*)(ws + 6194432);   // B               (32,768)
  float*    s    = (float*)(ws + 6227200);      // B               (32,768)
  bf16*     wb   = (bf16*)(ws + 6259968);       // 5*16384 bf16    (163,840)
  const bf16* wf_n1 = wb;
  const bf16* wf_n2 = wb + 16384;
  const bf16* wf_g1 = wb + 32768;
  const bf16* wf_q1 = wb + 49152;
  const bf16* wf_q2 = wb + 65536;

  k_init <<<(BB * CC + 255) / 256, 256, 0, stream>>>(out, menc, s);
  k_wconv<<<5, 256, 0, stream>>>(nw1, nw2, gw1, qw1, qw2, wb);
  k_qmlp <<<BB / 128, 256, 0, stream>>>(u, qb1, qb2, wf_q1, wf_q2, q);
  k_gate <<<NTILES, 256, 0, stream>>>(x, batch, q, nb1, nb2, gb1, gw2, gb2,
                                      wf_n1, wf_n2, wf_g1, ge, menc);
  k_soft <<<(NN + 255) / 256, 256, 0, stream>>>(batch, ge, menc, s);
  k_pool <<<NTILES, 256, 0, stream>>>(x, batch, nb1, nb2, wf_n1, wf_n2, ge, s, out);
}
